// BackBone_77532749627801
// MI455X (gfx1250) — matrix operand order repaired and checked by reading, not compiled, not run on hardware
//
#include <hip/hip_runtime.h>
#include <math.h>

// ---------------------------------------------------------------------------
// Transformer encoder layer for MI455X (gfx1250, wave32, WMMA bf16).
// B=4, S=2048, D=1024, H=16, DH=64, F=4096.  All GEMMs via
// v_wmma_f32_16x16x32_bf16, f32 accumulation, f32 layernorm/residual path.
// GEMM waves are register-blocked 2x4 (32x64 output per wave); flash
// attention stages shared K/V tiles into LDS once per block through the
// Tensor Data Mover (tensor_load_to_lds + s_wait_tensorcnt).
// ---------------------------------------------------------------------------

typedef __attribute__((ext_vector_type(16))) __bf16 v16bf;
typedef __attribute__((ext_vector_type(8)))  __bf16 v8bf;
typedef __attribute__((ext_vector_type(8)))  float  v8f;
typedef __attribute__((ext_vector_type(4)))  unsigned int u32x4;
typedef __attribute__((ext_vector_type(8)))  int i32x8;
typedef __attribute__((ext_vector_type(4)))  int i32x4;

#define CB 4
#define CS 2048
#define CD 1024
#define CH 16
#define CDH 64
#define CF 4096
#define CM (CB * CS)          // 8192 rows of tokens

__device__ __forceinline__ v8f zero8() {
  v8f z = {0.f, 0.f, 0.f, 0.f, 0.f, 0.f, 0.f, 0.f};
  return z;
}

__device__ __forceinline__ v8f wmma_bf16(v16bf a, v16bf b, v8f c) {
  // (neg_a, A, neg_b, B, c_mod, C, reuse_a, reuse_b)
  return __builtin_amdgcn_wmma_f32_16x16x32_bf16(false, a, false, b,
                                                 (short)0, c, false, false);
}

// A-matrix 16x32 tile (MxK) from row-major [ld >= 32] bf16:
// lanes 0-15: row = lane,   K = 0..15
// lanes16-31: row = lane-16,K = 16..31
__device__ __forceinline__ v16bf load_a_tile(const __bf16* __restrict__ base,
                                             int ld, int lane) {
  int r  = lane & 15;
  int kb = (lane >> 4) << 4;                 // 0 or 16
  const __bf16* p = base + (size_t)r * ld + kb;
  v8bf lo = *(const v8bf*)(p);               // K = kb .. kb+7
  v8bf hi = *(const v8bf*)(p + 8);           // K = kb+8 .. kb+15
  return __builtin_shufflevector(lo, hi, 0,1,2,3,4,5,6,7,8,9,10,11,12,13,14,15);
}

// B-matrix 32x16 tile (KxN) from Bt row-major [N][K] (ld = K stride):
// lanes 0-15: col = lane,   K = 0..15 ; lanes 16-31: col = lane-16, K = 16..31
__device__ __forceinline__ v16bf load_b_tile(const __bf16* __restrict__ base,
                                             int ld, int lane) {
  int c  = lane & 15;
  int kb = (lane >> 4) << 4;                 // 0 or 16
  const __bf16* p = base + (size_t)c * ld + kb;
  v8bf lo = *(const v8bf*)(p);
  v8bf hi = *(const v8bf*)(p + 8);
  return __builtin_shufflevector(lo, hi, 0,1,2,3,4,5,6,7,8,9,10,11,12,13,14,15);
}

// LDS variant of load_b_tile (ds_load path).
__device__ __forceinline__ v16bf load_b_tile_lds(const __bf16* base, int ld,
                                                 int lane) {
  int c  = lane & 15;
  int kb = (lane >> 4) << 4;
  const __bf16* p = base + (size_t)c * ld + kb;
  v8bf lo = *(const v8bf*)(p);
  v8bf hi = *(const v8bf*)(p + 8);
  return __builtin_shufflevector(lo, hi, 0,1,2,3,4,5,6,7,8,9,10,11,12,13,14,15);
}

// ---------------------------------------------------------------------------
// Tensor Data Mover: async 2D tile load (global -> LDS), bf16 elements.
// D# built per CDNA5 ISA; groups 2/3 zero (2D).
// 6-arg builtin form (clang-23 / therock HIP headers).
// ---------------------------------------------------------------------------
__device__ __forceinline__ void tdm_load_2d(unsigned int lds_addr,
                                            const void* gaddr,
                                            unsigned int tile_d0,      // elems/row
                                            unsigned int tile_d1,      // rows
                                            unsigned long long stride0, // elems
                                            unsigned int tensor_d0,
                                            unsigned int tensor_d1) {
  unsigned long long ga = (unsigned long long)(uintptr_t)gaddr;
  u32x4 g0;
  g0[0] = 1u;                                           // count=1
  g0[1] = lds_addr;                                     // lds_addr [63:32]
  g0[2] = (unsigned int)(ga & 0xffffffffu);             // global_addr [95:64]
  g0[3] = (unsigned int)((ga >> 32) & 0x01ffffffu) | (2u << 30); // [120:96] + type=2
  i32x8 g1;
  g1[0] = (int)(1u << 16);                              // data_size=1 (2B)
  g1[1] = (int)((tensor_d0 & 0xffffu) << 16);           // tensor_dim0[15:0]
  g1[2] = (int)(((tensor_d0 >> 16) & 0xffffu) |
                ((tensor_d1 & 0xffffu) << 16));         // dim0[31:16] | dim1[15:0]
  g1[3] = (int)(((tensor_d1 >> 16) & 0xffffu) |
                ((tile_d0 & 0xffffu) << 16));           // dim1[31:16] | tile_dim0
  g1[4] = (int)(tile_d1 & 0xffffu);                     // tile_dim1 (tile_dim2=0)
  g1[5] = (int)(unsigned int)(stride0 & 0xffffffffu);   // dim0_stride[31:0]
  g1[6] = (int)(unsigned int)((stride0 >> 32) & 0xffffu); // dim0_stride[47:32]
  g1[7] = 0;
  i32x4 z4 = {0, 0, 0, 0};
  i32x8 z8 = {0, 0, 0, 0, 0, 0, 0, 0};
  __builtin_amdgcn_tensor_load_to_lds(g0, g1, z4, z4, z8, 0);
}

// Register-blocked GEMM core: one wave accumulates a 32x64 output region
// (2 M-tiles x 4 N-tiles).  A row-major [M,K]; Bt row-major [N,K]; K % 32 == 0.
__device__ __forceinline__ void gemm_core_2x4(const __bf16* __restrict__ A0,
                                              const __bf16* __restrict__ Bt0,
                                              int K, int lane, v8f acc[2][4]) {
  const __bf16* A1 = A0 + (size_t)16 * K;
  const __bf16* B1 = Bt0 + (size_t)16 * K;
  const __bf16* B2 = Bt0 + (size_t)32 * K;
  const __bf16* B3 = Bt0 + (size_t)48 * K;
#pragma unroll 2
  for (int k = 0; k < K; k += 32) {
    if (k + 256 < K) {                       // global_prefetch_b8, 8 steps ahead
      __builtin_prefetch(A0 + k + 256, 0, 1);
      __builtin_prefetch(Bt0 + k + 256, 0, 1);
    }
    v16bf a0 = load_a_tile(A0 + k, K, lane);
    v16bf a1 = load_a_tile(A1 + k, K, lane);
    v16bf b0 = load_b_tile(Bt0 + k, K, lane);
    v16bf b1 = load_b_tile(B1 + k, K, lane);
    v16bf b2 = load_b_tile(B2 + k, K, lane);
    v16bf b3 = load_b_tile(B3 + k, K, lane);
    acc[0][0] = wmma_bf16(a0, b0, acc[0][0]);
    acc[0][1] = wmma_bf16(a0, b1, acc[0][1]);
    acc[0][2] = wmma_bf16(a0, b2, acc[0][2]);
    acc[0][3] = wmma_bf16(a0, b3, acc[0][3]);
    acc[1][0] = wmma_bf16(a1, b0, acc[1][0]);
    acc[1][1] = wmma_bf16(a1, b1, acc[1][1]);
    acc[1][2] = wmma_bf16(a1, b2, acc[1][2]);
    acc[1][3] = wmma_bf16(a1, b3, acc[1][3]);
  }
}

// ---------------------------------------------------------------------------
// Prep kernels: convert / transpose into bf16 Bt layouts
// ---------------------------------------------------------------------------
__global__ void cvt_f32_bf16_kernel(const float* __restrict__ in,
                                    __bf16* __restrict__ out, int n) {
  int i = blockIdx.x * 256 + threadIdx.x;
  if (i < n) out[i] = (__bf16)in[i];
}

// Wcat[n][d] with n = sec*1024 + h*64 + e, sec in {q,k,v}; src W[h][d][e].
__global__ void pack_wcat_kernel(const float* __restrict__ Wq,
                                 const float* __restrict__ Wk,
                                 const float* __restrict__ Wv,
                                 const float* __restrict__ bq,
                                 const float* __restrict__ bk,
                                 const float* __restrict__ bv,
                                 __bf16* __restrict__ Wcat,
                                 float* __restrict__ bcat) {
  int idx = blockIdx.x * 256 + threadIdx.x;
  if (idx >= 3 * CD * CD) return;
  int n = idx >> 10, d = idx & 1023;
  int sec = n >> 10, nn = n & 1023;
  int h = nn >> 6, e = nn & 63;
  const float* W = (sec == 0) ? Wq : (sec == 1) ? Wk : Wv;
  Wcat[idx] = (__bf16)W[((size_t)h * CD + d) * CDH + e];
  if (d == 0) {
    const float* bb = (sec == 0) ? bq : (sec == 1) ? bk : bv;
    bcat[n] = bb[h * CDH + e];
  }
}

// Wt[c][r] = W[r][c];  W is [R,C] row-major.
__global__ void transpose_w_kernel(const float* __restrict__ W,
                                   __bf16* __restrict__ Wt, int R, int C) {
  int idx = blockIdx.x * 256 + threadIdx.x;
  if (idx >= R * C) return;
  int c = idx / R, r = idx - c * R;
  Wt[idx] = (__bf16)W[(size_t)r * C + c];
}

// ---------------------------------------------------------------------------
// Fused QKV projection: [8192,1024] @ Wcat^T[1024,3072] + bias.
// q,k stored [BH, S, 64] bf16; v stored transposed [BH, 64, S] bf16.
// ---------------------------------------------------------------------------
__global__ void __launch_bounds__(256)
qkv_gemm_kernel(const __bf16* __restrict__ xb, const __bf16* __restrict__ Wcat,
                const float* __restrict__ bcat, __bf16* __restrict__ q,
                __bf16* __restrict__ kbuf, __bf16* __restrict__ vT) {
  int lane = threadIdx.x & 31, wave = threadIdx.x >> 5;
  int n0 = (blockIdx.x * 8 + wave) * 64;
  int m0 = blockIdx.y * 32;
  v8f acc[2][4];
#pragma unroll
  for (int i = 0; i < 2; ++i)
#pragma unroll
    for (int j = 0; j < 4; ++j) acc[i][j] = zero8();
  gemm_core_2x4(xb + (size_t)m0 * CD, Wcat + (size_t)n0 * CD, CD, lane, acc);

  int col = lane & 15, grp = lane >> 4;
#pragma unroll
  for (int nt = 0; nt < 4; ++nt) {
    int n = n0 + nt * 16 + col;
    float bias = bcat[n];
    int sec = n >> 10, nn = n & 1023, h = nn >> 6, e = nn & 63;
#pragma unroll
    for (int mt = 0; mt < 2; ++mt) {
#pragma unroll
      for (int v = 0; v < 8; ++v) {
        int m = m0 + mt * 16 + grp * 8 + v;
        int b = m >> 11, s = m & (CS - 1);
        __bf16 val = (__bf16)(acc[mt][nt][v] + bias);
        size_t bh = (size_t)(b * CH + h);
        if (sec == 0)      q[(bh * CS + s) * CDH + e]    = val;
        else if (sec == 1) kbuf[(bh * CS + s) * CDH + e] = val;
        else               vT[(bh * CDH + e) * CS + s]   = val;
      }
    }
  }
}

// ---------------------------------------------------------------------------
// Flash attention: block = 128 query rows of one (b,h); wave = 16 rows.
// K/V tiles (32 keys) staged into LDS once per block via the Tensor Data
// Mover; online softmax wave-local; P routed to A layout via LDS bounce.
// ---------------------------------------------------------------------------
__global__ void __launch_bounds__(256)
flash_attn_kernel(const __bf16* __restrict__ qb, const __bf16* __restrict__ kb,
                  const __bf16* __restrict__ vT, float* __restrict__ mha) {
  __shared__ __bf16 sK[32 * 64];       // [key 0..31][e 0..63]
  __shared__ __bf16 sV[64 * 32];       // [e 0..63][key 0..31]
  __shared__ float  sP[8][16 * 32];
  int lane = threadIdx.x & 31, wave = threadIdx.x >> 5;
  int grp = lane >> 4, col = lane & 15;
  int bh = blockIdx.y;
  int b = bh >> 4, h = bh & 15;
  int q0 = blockIdx.x * 128 + wave * 16;

  const __bf16* Q  = qb + ((size_t)bh * CS + q0) * CDH;
  const __bf16* Kp = kb + (size_t)bh * CS * CDH;   // [S,64] rows = keys
  const __bf16* Vp = vT + (size_t)bh * CDH * CS;   // [64,S] rows = e
  float* sw = sP[wave];
  unsigned int ldsK = (unsigned int)(uintptr_t)&sK[0];
  unsigned int ldsV = (unsigned int)(uintptr_t)&sV[0];

  v16bf a0 = load_a_tile(Q, CDH, lane);            // e = 0..31
  v16bf a1 = load_a_tile(Q + 32, CDH, lane);       // e = 32..63

  v8f o0 = zero8(), o1 = zero8(), o2 = zero8(), o3 = zero8();
  float mrow[8], lrow[8];
#pragma unroll
  for (int v = 0; v < 8; ++v) { mrow[v] = -1e30f; lrow[v] = 0.f; }

  const float scale = 0.125f;                      // 1/sqrt(64)
  for (int kt = 0; kt < CS; kt += 32) {
    __syncthreads();                               // prior reads done
    if (wave == 0) {
      // K tile: 32 rows (keys) x 64 elems, row stride 64.
      tdm_load_2d(ldsK, Kp + (size_t)kt * CDH, CDH, 32, CDH, CDH, CS - kt);
      __builtin_amdgcn_s_wait_tensorcnt(0);
    } else if (wave == 1) {
      // V tile: 64 rows (e) x 32 elems, row stride S.
      tdm_load_2d(ldsV, Vp + kt, 32, CDH, CS, CS - kt, CDH);
      __builtin_amdgcn_s_wait_tensorcnt(0);
    }
    __syncthreads();                               // tiles visible

    v16bf b00 = load_b_tile_lds(sK, CDH, lane);             // keys 0..15, e 0..31
    v16bf b01 = load_b_tile_lds(sK + 32, CDH, lane);        // keys 0..15, e 32..63
    v16bf b10 = load_b_tile_lds(sK + 16 * CDH, CDH, lane);  // keys 16..31
    v16bf b11 = load_b_tile_lds(sK + 16 * CDH + 32, CDH, lane);
    v8f s0 = zero8(), s1 = zero8();
    s0 = wmma_bf16(a0, b00, s0); s0 = wmma_bf16(a1, b01, s0);
    s1 = wmma_bf16(a0, b10, s1); s1 = wmma_bf16(a1, b11, s1);

#pragma unroll
    for (int v = 0; v < 8; ++v) {
      float x0 = s0[v] * scale, x1 = s1[v] * scale;
      float t = fmaxf(x0, x1);
      t = fmaxf(t, __shfl_xor(t, 1)); t = fmaxf(t, __shfl_xor(t, 2));
      t = fmaxf(t, __shfl_xor(t, 4)); t = fmaxf(t, __shfl_xor(t, 8));
      float mn = fmaxf(mrow[v], t);
      float corr = __expf(mrow[v] - mn);
      float p0 = __expf(x0 - mn), p1 = __expf(x1 - mn);
      float ps = p0 + p1;
      ps += __shfl_xor(ps, 1); ps += __shfl_xor(ps, 2);
      ps += __shfl_xor(ps, 4); ps += __shfl_xor(ps, 8);
      lrow[v] = lrow[v] * corr + ps;
      mrow[v] = mn;
      o0[v] *= corr; o1[v] *= corr; o2[v] *= corr; o3[v] *= corr;
      int r = v + grp * 8;                          // query row in tile
      sw[r * 32 + col]      = p0;                   // key cols 0..15
      sw[r * 32 + 16 + col] = p1;                   // key cols 16..31
    }
    // Re-read P in A-matrix layout (wave-private LDS region; DS ops are
    // in-order per wave, compiler inserts s_wait_dscnt before use).
    int kb2 = grp * 16;
    v16bf pa;
#pragma unroll
    for (int i = 0; i < 8; ++i) {
      pa[i]     = (__bf16)sw[col * 32 + kb2 + i];
      pa[i + 8] = (__bf16)sw[col * 32 + kb2 + 8 + i];
    }
    v16bf pv0 = load_b_tile_lds(sV, 32, lane);
    v16bf pv1 = load_b_tile_lds(sV + 16 * 32, 32, lane);
    v16bf pv2 = load_b_tile_lds(sV + 32 * 32, 32, lane);
    v16bf pv3 = load_b_tile_lds(sV + 48 * 32, 32, lane);
    o0 = wmma_bf16(pa, pv0, o0); o1 = wmma_bf16(pa, pv1, o1);
    o2 = wmma_bf16(pa, pv2, o2); o3 = wmma_bf16(pa, pv3, o3);
  }

#pragma unroll
  for (int v = 0; v < 8; ++v) {
    float inv = 1.0f / lrow[v];
    int s = q0 + grp * 8 + v;
    size_t base = ((size_t)b * CS + s) * CD + h * CDH + col;
    mha[base +  0] = o0[v] * inv;
    mha[base + 16] = o1[v] * inv;
    mha[base + 32] = o2[v] * inv;
    mha[base + 48] = o3[v] * inv;
  }
}

// ---------------------------------------------------------------------------
// AddNorm: out = LN(A (+ R)) * g + b ; one block per token row (D=1024).
// ---------------------------------------------------------------------------
__global__ void __launch_bounds__(256)
addnorm_kernel(const float* __restrict__ A, const float* __restrict__ R,
               const float* __restrict__ g, const float* __restrict__ bb,
               float* __restrict__ outF, __bf16* __restrict__ outB) {
  int row = blockIdx.x, tid = threadIdx.x;
  int lane = tid & 31, wave = tid >> 5;
  size_t base = (size_t)row * CD;
  float vals[4], s = 0.f, ss = 0.f;
#pragma unroll
  for (int i = 0; i < 4; ++i) {
    int d = tid + i * 256;
    float v = A[base + d];
    if (R) v += R[base + d];
    vals[i] = v; s += v; ss += v * v;
  }
  s += __shfl_xor(s, 16); ss += __shfl_xor(ss, 16);
  s += __shfl_xor(s, 8);  ss += __shfl_xor(ss, 8);
  s += __shfl_xor(s, 4);  ss += __shfl_xor(ss, 4);
  s += __shfl_xor(s, 2);  ss += __shfl_xor(ss, 2);
  s += __shfl_xor(s, 1);  ss += __shfl_xor(ss, 1);
  __shared__ float rs[8], rq[8];
  if (lane == 0) { rs[wave] = s; rq[wave] = ss; }
  __syncthreads();
  s = 0.f; ss = 0.f;
#pragma unroll
  for (int w = 0; w < 8; ++w) { s += rs[w]; ss += rq[w]; }
  float mean = s * (1.0f / CD);
  float var  = ss * (1.0f / CD) - mean * mean;
  float rstd = rsqrtf(var + 1e-5f);
#pragma unroll
  for (int i = 0; i < 4; ++i) {
    int d = tid + i * 256;
    float o = (vals[i] - mean) * rstd * g[d] + bb[d];
    outF[base + d] = o;
    if (outB) outB[base + d] = (__bf16)o;
  }
}

// ---------------------------------------------------------------------------
// FFN GEMMs (2x4 register-blocked waves)
// ---------------------------------------------------------------------------
__global__ void __launch_bounds__(256)
ffn1_kernel(const __bf16* __restrict__ h1b, const __bf16* __restrict__ W1t,
            const float* __restrict__ b1, __bf16* __restrict__ ffb) {
  int lane = threadIdx.x & 31, wave = threadIdx.x >> 5;
  int n0 = (blockIdx.x * 8 + wave) * 64;
  int m0 = blockIdx.y * 32;
  v8f acc[2][4];
#pragma unroll
  for (int i = 0; i < 2; ++i)
#pragma unroll
    for (int j = 0; j < 4; ++j) acc[i][j] = zero8();
  gemm_core_2x4(h1b + (size_t)m0 * CD, W1t + (size_t)n0 * CD, CD, lane, acc);

  int col = lane & 15, grp = lane >> 4;
#pragma unroll
  for (int nt = 0; nt < 4; ++nt) {
    int n = n0 + nt * 16 + col;
    float bias = b1[n];
#pragma unroll
    for (int mt = 0; mt < 2; ++mt) {
#pragma unroll
      for (int v = 0; v < 8; ++v) {
        int m = m0 + mt * 16 + grp * 8 + v;
        float x = acc[mt][nt][v] + bias;
        float ge = 0.5f * x * (1.0f + erff(x * 0.70710678118654752f));
        ffb[(size_t)m * CF + n] = (__bf16)ge;
      }
    }
  }
}

__global__ void __launch_bounds__(256)
ffn2_kernel(const __bf16* __restrict__ ffb, const __bf16* __restrict__ W2t,
            const float* __restrict__ b2, const float* __restrict__ h1f,
            float* __restrict__ y) {
  int lane = threadIdx.x & 31, wave = threadIdx.x >> 5;
  int n0 = (blockIdx.x * 8 + wave) * 64;
  int m0 = blockIdx.y * 32;
  v8f acc[2][4];
#pragma unroll
  for (int i = 0; i < 2; ++i)
#pragma unroll
    for (int j = 0; j < 4; ++j) acc[i][j] = zero8();
  gemm_core_2x4(ffb + (size_t)m0 * CF, W2t + (size_t)n0 * CF, CF, lane, acc);

  int col = lane & 15, grp = lane >> 4;
#pragma unroll
  for (int nt = 0; nt < 4; ++nt) {
    int n = n0 + nt * 16 + col;
    float bias = b2[n];
#pragma unroll
    for (int mt = 0; mt < 2; ++mt) {
#pragma unroll
      for (int v = 0; v < 8; ++v) {
        int m = m0 + mt * 16 + grp * 8 + v;
        size_t idx = (size_t)m * CD + n;
        y[idx] = acc[mt][nt][v] + bias + h1f[idx];   // residual fused
      }
    }
  }
}

// ---------------------------------------------------------------------------
// Launch
// ---------------------------------------------------------------------------
extern "C" void kernel_launch(void* const* d_in, const int* in_sizes, int n_in,
                              void* d_out, int out_size, void* d_ws, size_t ws_size,
                              hipStream_t stream) {
  const float* x    = (const float*)d_in[0];
  const float* Wq   = (const float*)d_in[1];
  const float* bq   = (const float*)d_in[2];
  const float* Wk   = (const float*)d_in[3];
  const float* bk   = (const float*)d_in[4];
  const float* Wv   = (const float*)d_in[5];
  const float* bv   = (const float*)d_in[6];
  const float* ln1g = (const float*)d_in[7];
  const float* ln1b = (const float*)d_in[8];
  const float* W1   = (const float*)d_in[9];
  const float* b1   = (const float*)d_in[10];
  const float* W2   = (const float*)d_in[11];
  const float* b2   = (const float*)d_in[12];
  const float* ln2g = (const float*)d_in[13];
  const float* ln2b = (const float*)d_in[14];
  float* out = (float*)d_out;

  char* ws = (char*)d_ws;
  size_t off = 0;
  auto carve = [&](size_t bytes) -> char* {
    char* p = ws + off;
    off += (bytes + 255) & ~(size_t)255;
    return p;
  };
  __bf16* xb   = (__bf16*)carve((size_t)CM * CD * 2);        // 16 MB
  __bf16* Wcat = (__bf16*)carve((size_t)3 * CD * CD * 2);    //  6 MB
  float*  bcat = (float*) carve((size_t)3 * CD * 4);
  __bf16* W1t  = (__bf16*)carve((size_t)CD * CF * 2);        //  8 MB
  __bf16* W2t  = (__bf16*)carve((size_t)CF * CD * 2);        //  8 MB
  __bf16* qb   = (__bf16*)carve((size_t)CM * CD * 2);        // 16 MB
  __bf16* kbuf = (__bf16*)carve((size_t)CM * CD * 2);        // 16 MB
  __bf16* vTb  = (__bf16*)carve((size_t)CM * CD * 2);        // 16 MB
  float*  mha  = (float*) carve((size_t)CM * CD * 4);        // 32 MB
  float*  h1f  = (float*) carve((size_t)CM * CD * 4);        // 32 MB
  __bf16* h1b  = (__bf16*)carve((size_t)CM * CD * 2);        // 16 MB
  __bf16* ffb  = (__bf16*)carve((size_t)CM * CF * 2);        // 64 MB
  float*  yb   = (float*) carve((size_t)CM * CD * 4);        // 32 MB

  // 1) Prep: bf16 conversions / Bt-layout weight packs.
  {
    int n = CM * CD;
    cvt_f32_bf16_kernel<<<(n + 255) / 256, 256, 0, stream>>>(x, xb, n);
  }
  pack_wcat_kernel<<<(3 * CD * CD + 255) / 256, 256, 0, stream>>>(
      Wq, Wk, Wv, bq, bk, bv, Wcat, bcat);
  transpose_w_kernel<<<(CD * CF + 255) / 256, 256, 0, stream>>>(W1, W1t, CD, CF);
  transpose_w_kernel<<<(CF * CD + 255) / 256, 256, 0, stream>>>(W2, W2t, CF, CD);

  // 2) Fused QKV projection: block covers 512 N cols x 32 M rows.
  qkv_gemm_kernel<<<dim3(3 * CD / 512, CM / 32), 256, 0, stream>>>(
      xb, Wcat, bcat, qb, kbuf, vTb);

  // 3) Flash attention: grid (S/128 query tiles, B*H heads).
  flash_attn_kernel<<<dim3(CS / 128, CB * CH), 256, 0, stream>>>(qb, kbuf, vTb, mha);

  // 4) AddNorm1 -> h1 (f32 + bf16).
  addnorm_kernel<<<CM, 256, 0, stream>>>(x, mha, ln1g, ln1b, h1f, h1b);

  // 5) FFN1 + exact GELU -> bf16.
  ffn1_kernel<<<dim3(CF / 512, CM / 32), 256, 0, stream>>>(h1b, W1t, b1, ffb);

  // 6) FFN2 + bias + residual -> f32.
  ffn2_kernel<<<dim3(CD / 512, CM / 32), 256, 0, stream>>>(ffb, W2t, b2, h1f, yb);

  // 7) AddNorm2 -> d_out.
  addnorm_kernel<<<CM, 256, 0, stream>>>(yb, nullptr, ln2g, ln2b, out, nullptr);
}
